// DeepPnP_19413252178441
// MI455X (gfx1250) — matrix operand order repaired and checked by reading, not compiled, not run on hardware
//
#include <hip/hip_runtime.h>
#include <hip/hip_bf16.h>
#include <math.h>

// ---------------------------------------------------------------------------
// CDNA5 (gfx1250) wave32 WMMA types + helpers
// ---------------------------------------------------------------------------
typedef __attribute__((ext_vector_type(16))) __bf16 v16bf;
typedef __attribute__((ext_vector_type(8)))  float  v8f;

// native hardware f32->bf16 convert (RNE) on gfx1250
static __device__ __forceinline__ __bf16 f2bf(float f) { return (__bf16)f; }
static __device__ __forceinline__ unsigned short bf_bits(float f) {
    return __builtin_bit_cast(unsigned short, (__bf16)f);
}

// CDNA5 async global->LDS copies (GLOBAL_LOAD_ASYNC_TO_LDS_B*, ASYNCcnt).
// Generic pointers to __shared__ carry the LDS byte offset in the low 32 bits
// which is what VDST wants.
static __device__ __forceinline__ void async_g2l4(float* lds_dst, const float* gsrc) {
    unsigned loff = (unsigned)(unsigned long long)lds_dst;
    unsigned long long ga = (unsigned long long)gsrc;
    asm volatile("global_load_async_to_lds_b128 %0, %1, off"
                 :: "v"(loff), "v"(ga) : "memory");
}
static __device__ __forceinline__ void async_wait() {
    asm volatile("s_wait_asynccnt 0" ::: "memory");
}

#define NPTS 1024
#define DCH  128
#define BSZ  4

// ---------------------------------------------------------------------------
// 1) normalize p2d -> f2  ([B,1024,3] point-major)
// ---------------------------------------------------------------------------
__global__ void normalize_pts(const float* __restrict__ p2d, float* __restrict__ f2) {
    int t = blockIdx.x * blockDim.x + threadIdx.x;
    if (t >= BSZ * NPTS) return;
    float x = p2d[t * 3 + 0], y = p2d[t * 3 + 1], z = p2d[t * 3 + 2];
    float inv = rsqrtf(x * x + y * y + z * z);
    f2[t * 3 + 0] = x * inv;
    f2[t * 3 + 1] = y * inv;
    f2[t * 3 + 2] = z * inv;
}

// ---------------------------------------------------------------------------
// 2) KNN (k=10) graph feature + encoder conv (mean over k folded through the
//    linear layer). pts: [B,1024,3]; desc out: [B,128,1024] channel-major.
// ---------------------------------------------------------------------------
__global__ void knn_encode(const float* __restrict__ pts,
                           const float* __restrict__ Wenc,
                           const float* __restrict__ benc,
                           float* __restrict__ desc) {
    int t = blockIdx.x * blockDim.x + threadIdx.x;
    if (t >= BSZ * NPTS) return;
    int b = t >> 10, n = t & 1023;
    const float* P = pts + (long)b * NPTS * 3;
    float cx = P[n * 3], cy = P[n * 3 + 1], cz = P[n * 3 + 2];

    float bd[10];
    int   bi[10];
#pragma unroll
    for (int j = 0; j < 10; ++j) { bd[j] = -1e30f; bi[j] = 0; }
    for (int m = 0; m < NPTS; ++m) {
        float dx = P[m * 3] - cx, dy = P[m * 3 + 1] - cy, dz = P[m * 3 + 2] - cz;
        float d = -(dx * dx + dy * dy + dz * dz);
        if (d > bd[9]) {
            int j = 9;
            while (j > 0 && bd[j - 1] < d) { bd[j] = bd[j - 1]; bi[j] = bi[j - 1]; --j; }
            bd[j] = d; bi[j] = m;
        }
    }
    float sx = 0.f, sy = 0.f, sz = 0.f;
#pragma unroll
    for (int j = 0; j < 10; ++j) {
        sx += P[bi[j] * 3]; sy += P[bi[j] * 3 + 1]; sz += P[bi[j] * 3 + 2];
    }
    const float inv = 0.1f;
    float feat[6] = { sx * inv - cx, sy * inv - cy, sz * inv - cz, cx, cy, cz };
    float* out = desc + (long)b * DCH * NPTS + n;
    for (int o = 0; o < DCH; ++o) {
        float acc = benc[o];
#pragma unroll
        for (int c = 0; c < 6; ++c) acc += Wenc[o * 6 + c] * feat[c];
        out[(long)o * NPTS] = acc;
    }
}

// ---------------------------------------------------------------------------
// 3) bf16 WMMA GEMM, 4 waves/block, LDS-staged via async b128 global->LDS.
//    Out[b][row+rowOff, n] = W[M,K] @ X[b][K,N=1024] + bias.
//    Block tile: 16 (M) x 64 (N); wave w owns N subtile w*16..w*16+15.
// ---------------------------------------------------------------------------
__global__ void wmma_gemm(const float* __restrict__ W, const float* __restrict__ X,
                          const float* __restrict__ bias, float* __restrict__ Out,
                          int Kdim, long xBatchStride, long oBatchStride, int oRowOff) {
    __shared__ float Ws[16][32];   // A slab: 16 rows x 32 K
    __shared__ float Xs[32][64];   // B slab: 32 K x 64 N
    const int tid = threadIdx.x;
    const int wid = tid >> 5, lane = tid & 31;
    const int lo = lane & 15, hi = lane >> 4;
    const int tn0 = blockIdx.x * 64, tm = blockIdx.y * 16;
    const float* Xb = X + (long)blockIdx.z * xBatchStride;
    float* Ob = Out + (long)blockIdx.z * oBatchStride;

    v8f acc = {};
    for (int k0 = 0; k0 < Kdim; k0 += 32) {
        // stage W tile: 512 floats = 128 b128 ops (1/thread), contiguous in K
        {
            int m = tid >> 3, k = (tid & 7) * 4;
            async_g2l4(&Ws[m][k], &W[(long)(tm + m) * Kdim + k0 + k]);
        }
        // stage X tile: 2048 floats = 512 b128 ops (4/thread), contiguous in N
#pragma unroll
        for (int i = 0; i < 4; ++i) {
            int fi = (i * 128 + tid) * 4;
            int k = fi >> 6, n = fi & 63;
            async_g2l4(&Xs[k][n], &Xb[(long)(k0 + k) * NPTS + tn0 + n]);
        }
        async_wait();
        __syncthreads();

        v16bf a, bfr;
#pragma unroll
        for (int j = 0; j < 8; ++j) {
            a[j]     = f2bf(Ws[lo][8 * hi + j]);        // K = 8*hi + j
            a[j + 8] = f2bf(Ws[lo][16 + 8 * hi + j]);   // K = 16 + 8*hi + j
        }
#pragma unroll
        for (int j = 0; j < 16; ++j)
            bfr[j] = f2bf(Xs[(j & 7) + 8 * hi + 16 * (j >> 3)][wid * 16 + lo]);
        acc = __builtin_amdgcn_wmma_f32_16x16x32_bf16(false, a, false, bfr,
                                                      (short)0, acc, false, false);
        __syncthreads();
    }
    const int col = tn0 + wid * 16 + lo;
#pragma unroll
    for (int r = 0; r < 8; ++r) {
        int row = tm + r + 8 * hi;
        Ob[(long)(row + oRowOff) * NPTS + col] = acc[r] + bias[row];
    }
}

// ---------------------------------------------------------------------------
// 4) Flash attention: 4 waves/block (4 query tiles) share async-staged 32x32
//    K and V blocks in LDS.  Q,K,V: [B,128,1024]; head h channels c = d*4+h.
// ---------------------------------------------------------------------------
__global__ void flash_attn(const float* __restrict__ Q, const float* __restrict__ K,
                           const float* __restrict__ V, float* __restrict__ O) {
    __shared__ float Ks[32][32];              // [d][m_local]
    __shared__ float Vs[32][32];              // [d][m_local]
    __shared__ unsigned short Pl[4][16][32];  // per-wave P tile (bf16 bits)

    const int tid = threadIdx.x;
    const int wid = tid >> 5, lane = tid & 31;
    const int lo = lane & 15, hi = lane >> 4;
    const int n0 = blockIdx.x * 64 + wid * 16;
    const int h = blockIdx.y;
    const long base = (long)blockIdx.z * DCH * NPTS;
    const float scale = 0.17677669529663687f;  // 1/sqrt(32)

    // Q fragment (A layout), scale folded in
    v16bf qa;
#pragma unroll
    for (int j = 0; j < 8; ++j) {
        int d0 = j + 8 * hi, d1 = 16 + j + 8 * hi;
        qa[j]     = f2bf(Q[base + (long)(d0 * 4 + h) * NPTS + n0 + lo] * scale);
        qa[j + 8] = f2bf(Q[base + (long)(d1 * 4 + h) * NPTS + n0 + lo] * scale);
    }

    v8f acc0 = {}, acc1 = {};
    float mrun[8], lrun[8];
#pragma unroll
    for (int r = 0; r < 8; ++r) { mrun[r] = -1e30f; lrun[r] = 0.f; }

    for (int mb = 0; mb < 32; ++mb) {
        const int m0 = mb * 32;
        // async-stage K and V 32x32 blocks (b128, contiguous along m)
#pragma unroll
        for (int i = 0; i < 2; ++i) {
            int fi = (i * 128 + tid) * 4;
            int d = fi >> 5, m = fi & 31;
            const long coff = base + (long)(d * 4 + h) * NPTS + m0 + m;
            async_g2l4(&Ks[d][m], &K[coff]);
            async_g2l4(&Vs[d][m], &V[coff]);
        }
        async_wait();
        __syncthreads();

        // score tiles S = Q^T K for two 16-wide key tiles
        v16bf kb;
        v8f z = {};
#pragma unroll
        for (int j = 0; j < 16; ++j) kb[j] = f2bf(Ks[(j & 7) + 8 * hi + 16 * (j >> 3)][lo]);
        v8f sL = __builtin_amdgcn_wmma_f32_16x16x32_bf16(false, qa, false, kb,
                                                         (short)0, z, false, false);
#pragma unroll
        for (int j = 0; j < 16; ++j) kb[j] = f2bf(Ks[(j & 7) + 8 * hi + 16 * (j >> 3)][16 + lo]);
        v8f sR = __builtin_amdgcn_wmma_f32_16x16x32_bf16(false, qa, false, kb,
                                                         (short)0, z, false, false);
        // online softmax per row (row M = r + 8*hi, spread over 16 lanes)
#pragma unroll
        for (int r = 0; r < 8; ++r) {
            float mx = fmaxf(sL[r], sR[r]);
#pragma unroll
            for (int off = 1; off < 16; off <<= 1) mx = fmaxf(mx, __shfl_xor(mx, off, 32));
            float mnew = fmaxf(mrun[r], mx);
            float corr = __expf(mrun[r] - mnew);
            acc0[r] *= corr; acc1[r] *= corr;
            float pL = __expf(sL[r] - mnew);
            float pR = __expf(sR[r] - mnew);
            float rs = pL + pR;
#pragma unroll
            for (int off = 1; off < 16; off <<= 1) rs += __shfl_xor(rs, off, 32);
            lrun[r] = lrun[r] * corr + rs;
            mrun[r] = mnew;
            Pl[wid][r + 8 * hi][lo]      = bf_bits(pL);
            Pl[wid][r + 8 * hi][16 + lo] = bf_bits(pR);
        }
        // P (A layout from per-wave LDS) @ V (B layout: row=m, col=d -> Vs[d][m])
        v16bf pa;
#pragma unroll
        for (int j = 0; j < 8; ++j) {
            pa[j]     = __builtin_bit_cast(__bf16, Pl[wid][lo][j + 8 * hi]);
            pa[j + 8] = __builtin_bit_cast(__bf16, Pl[wid][lo][16 + j + 8 * hi]);
        }
        v16bf vb;
#pragma unroll
        for (int j = 0; j < 16; ++j) vb[j] = f2bf(Vs[lo][(j & 7) + 8 * hi + 16 * (j >> 3)]);
        acc0 = __builtin_amdgcn_wmma_f32_16x16x32_bf16(false, pa, false, vb,
                                                       (short)0, acc0, false, false);
#pragma unroll
        for (int j = 0; j < 16; ++j) vb[j] = f2bf(Vs[16 + lo][(j & 7) + 8 * hi + 16 * (j >> 3)]);
        acc1 = __builtin_amdgcn_wmma_f32_16x16x32_bf16(false, pa, false, vb,
                                                       (short)0, acc1, false, false);
        __syncthreads();
    }
    // write out: channel = d*4 + h
#pragma unroll
    for (int r = 0; r < 8; ++r) {
        float inv = 1.f / lrun[r];
        int n = n0 + r + 8 * hi;
        O[base + (long)(lo * 4 + h) * NPTS + n]        = acc0[r] * inv;
        O[base + (long)((16 + lo) * 4 + h) * NPTS + n] = acc1[r] * inv;
    }
}

// ---------------------------------------------------------------------------
// 5) concat lower half: ycat[:, :128, :] = x
// ---------------------------------------------------------------------------
__global__ void concat_lower(const float* __restrict__ x, float* __restrict__ ycat) {
    long i = (long)blockIdx.x * blockDim.x + threadIdx.x;
    if (i >= (long)BSZ * DCH * NPTS) return;
    long b = i >> 17, r = i & 131071;
    ycat[b * 262144 + r] = x[i];
}

// ---------------------------------------------------------------------------
// 6) BN stats over (B, N) per channel  (z: [B,256,1024])
// ---------------------------------------------------------------------------
__global__ void bn_stats(const float* __restrict__ z, float* __restrict__ mean,
                         float* __restrict__ var) {
    int c = blockIdx.x, tid = threadIdx.x;
    float s = 0.f, s2 = 0.f;
    for (int b = 0; b < BSZ; ++b) {
        const float* p = z + (long)b * 262144 + (long)c * NPTS;
        for (int n = tid; n < NPTS; n += 256) { float v = p[n]; s += v; s2 += v * v; }
    }
    __shared__ float sh[256], sh2[256];
    sh[tid] = s; sh2[tid] = s2; __syncthreads();
    for (int o = 128; o > 0; o >>= 1) {
        if (tid < o) { sh[tid] += sh[tid + o]; sh2[tid] += sh2[tid + o]; }
        __syncthreads();
    }
    if (tid == 0) {
        float m = sh[0] * (1.f / 4096.f);
        mean[c] = m;
        var[c] = sh2[0] * (1.f / 4096.f) - m * m;
    }
}

__global__ void bn_apply_relu(float* __restrict__ z, const float* __restrict__ g,
                              const float* __restrict__ bt, const float* __restrict__ mean,
                              const float* __restrict__ var) {
    long i = (long)blockIdx.x * blockDim.x + threadIdx.x;
    if (i >= (long)BSZ * 256 * NPTS) return;
    int c = (int)((i >> 10) & 255);
    float v = (z[i] - mean[c]) * rsqrtf(var[c] + 1e-5f) * g[c] + bt[c];
    z[i] = fmaxf(v, 0.f);
}

// ---------------------------------------------------------------------------
// 7) residual add
// ---------------------------------------------------------------------------
__global__ void add_res(const float* __restrict__ a, const float* __restrict__ d,
                        float* __restrict__ o, long n) {
    long i = (long)blockIdx.x * blockDim.x + threadIdx.x;
    if (i < n) o[i] = a[i] + d[i];
}

// ---------------------------------------------------------------------------
// 8) squared norms per point:  nrm[b,m] = sum_d desc[b,d,m]^2
// ---------------------------------------------------------------------------
__global__ void point_norms(const float* __restrict__ d, float* __restrict__ nrm) {
    int t = blockIdx.x * blockDim.x + threadIdx.x;
    if (t >= BSZ * NPTS) return;
    int b = t >> 10, m = t & 1023;
    const float* p = d + (long)b * DCH * NPTS + m;
    float s = 0.f;
    for (int c = 0; c < DCH; ++c) { float v = p[(long)c * NPTS]; s += v * v; }
    nrm[t] = s;
}

// ---------------------------------------------------------------------------
// 9) correlation GEMM (A^T B, K=128), async b128/LDS-staged, 4 waves/block,
//    with -sqrt(clip(dist2)) epilogue -> log_alpha
// ---------------------------------------------------------------------------
__global__ void wmma_corr(const float* __restrict__ D0, const float* __restrict__ D1,
                          const float* __restrict__ n0, const float* __restrict__ n1,
                          float* __restrict__ loga) {
    __shared__ float As[32][16];   // [k][m]  (global contiguity is along m)
    __shared__ float Bs[32][64];   // [k][n]
    const int tid = threadIdx.x;
    const int wid = tid >> 5, lane = tid & 31;
    const int lo = lane & 15, hi = lane >> 4;
    const int tn0 = blockIdx.x * 64, tm = blockIdx.y * 16;
    const long base = (long)blockIdx.z * DCH * NPTS;

    v8f acc = {};
    for (int k0 = 0; k0 < DCH; k0 += 32) {
        // A slab: 512 floats = 128 b128 ops (1/thread), contiguous along m
        {
            int k = tid >> 2, m = (tid & 3) * 4;
            async_g2l4(&As[k][m], &D0[base + (long)(k0 + k) * NPTS + tm + m]);
        }
        // B slab: 2048 floats = 512 b128 ops (4/thread), contiguous along n
#pragma unroll
        for (int i = 0; i < 4; ++i) {
            int fi = (i * 128 + tid) * 4;
            int k = fi >> 6, n = fi & 63;
            async_g2l4(&Bs[k][n], &D1[base + (long)(k0 + k) * NPTS + tn0 + n]);
        }
        async_wait();
        __syncthreads();

        v16bf a, bfr;
#pragma unroll
        for (int j = 0; j < 8; ++j) {
            a[j]     = f2bf(As[8 * hi + j][lo]);
            a[j + 8] = f2bf(As[16 + 8 * hi + j][lo]);
        }
#pragma unroll
        for (int j = 0; j < 16; ++j)
            bfr[j] = f2bf(Bs[(j & 7) + 8 * hi + 16 * (j >> 3)][wid * 16 + lo]);
        acc = __builtin_amdgcn_wmma_f32_16x16x32_bf16(false, a, false, bfr,
                                                      (short)0, acc, false, false);
        __syncthreads();
    }
    const int col = tn0 + wid * 16 + lo;
#pragma unroll
    for (int r = 0; r < 8; ++r) {
        int row = tm + r + 8 * hi;
        float d2 = n0[blockIdx.z * NPTS + row] + n1[blockIdx.z * NPTS + col] - 2.f * acc[r];
        d2 = fmaxf(d2, 1e-30f);
        loga[(long)blockIdx.z * NPTS * NPTS + (long)row * NPTS + col] = -sqrtf(d2);
    }
}

// ---------------------------------------------------------------------------
// 10) Sinkhorn log-sum-exp normalizations (in place)
// ---------------------------------------------------------------------------
__global__ void lse_rows(float* __restrict__ A) {
    float* row = A + (long)blockIdx.x * NPTS;
    int tid = threadIdx.x;
    __shared__ float sh[256];
    float mx = -1e30f;
    for (int i = tid; i < NPTS; i += 256) mx = fmaxf(mx, row[i]);
    sh[tid] = mx; __syncthreads();
    for (int o = 128; o > 0; o >>= 1) { if (tid < o) sh[tid] = fmaxf(sh[tid], sh[tid + o]); __syncthreads(); }
    mx = sh[0]; __syncthreads();
    float s = 0.f;
    for (int i = tid; i < NPTS; i += 256) s += __expf(row[i] - mx);
    sh[tid] = s; __syncthreads();
    for (int o = 128; o > 0; o >>= 1) { if (tid < o) sh[tid] += sh[tid + o]; __syncthreads(); }
    float lse = mx + __logf(sh[0]);
    for (int i = tid; i < NPTS; i += 256) row[i] -= lse;
}

__global__ void lse_cols(float* __restrict__ A) {
    int b = blockIdx.x >> 10, c = blockIdx.x & 1023;
    float* col = A + (long)b * NPTS * NPTS + c;
    int tid = threadIdx.x;
    __shared__ float sh[256];
    float mx = -1e30f;
    for (int i = tid; i < NPTS; i += 256) mx = fmaxf(mx, col[(long)i * NPTS]);
    sh[tid] = mx; __syncthreads();
    for (int o = 128; o > 0; o >>= 1) { if (tid < o) sh[tid] = fmaxf(sh[tid], sh[tid + o]); __syncthreads(); }
    mx = sh[0]; __syncthreads();
    float s = 0.f;
    for (int i = tid; i < NPTS; i += 256) s += __expf(col[(long)i * NPTS] - mx);
    sh[tid] = s; __syncthreads();
    for (int o = 128; o > 0; o >>= 1) { if (tid < o) sh[tid] += sh[tid + o]; __syncthreads(); }
    float lse = mx + __logf(sh[0]);
    for (int i = tid; i < NPTS; i += 256) col[(long)i * NPTS] -= lse;
}

// ---------------------------------------------------------------------------
// 11) Rodrigues + transform + renormalize
// ---------------------------------------------------------------------------
__global__ void transform_points(const float* __restrict__ pose, const float* __restrict__ p3d,
                                 float* __restrict__ p3dt) {
    int b = blockIdx.x;
    __shared__ float R[9], T[3];
    if (threadIdx.x == 0) {
        float ax = pose[b * 6], ay = pose[b * 6 + 1], az = pose[b * 6 + 2];
        float th = fmaxf(sqrtf(ax * ax + ay * ay + az * az), 1e-8f);
        float rx = ax / th, ry = ay / th, rz = az / th;
        float c = cosf(th), s = sinf(th), ic = 1.f - c;
        R[0] = c + ic * rx * rx;      R[1] = ic * rx * ry - s * rz; R[2] = ic * rx * rz + s * ry;
        R[3] = ic * ry * rx + s * rz; R[4] = c + ic * ry * ry;      R[5] = ic * ry * rz - s * rx;
        R[6] = ic * rz * rx - s * ry; R[7] = ic * rz * ry + s * rx; R[8] = c + ic * rz * rz;
        T[0] = pose[b * 6 + 3]; T[1] = pose[b * 6 + 4]; T[2] = pose[b * 6 + 5];
    }
    __syncthreads();
    for (int n = threadIdx.x; n < NPTS; n += blockDim.x) {
        const float* p = p3d + (long)b * NPTS * 3 + n * 3;
        float x = p[0], y = p[1], z = p[2];
        float ox = R[0] * x + R[1] * y + R[2] * z + T[0];
        float oy = R[3] * x + R[4] * y + R[5] * z + T[1];
        float oz = R[6] * x + R[7] * y + R[8] * z + T[2];
        float inv = rsqrtf(ox * ox + oy * oy + oz * oz);
        float* o = p3dt + (long)b * NPTS * 3 + n * 3;
        o[0] = ox * inv; o[1] = oy * inv; o[2] = oz * inv;
    }
}

// ---------------------------------------------------------------------------
// 12) final weighted error reduction
// ---------------------------------------------------------------------------
__global__ void zero_out(float* __restrict__ out) {
    if (threadIdx.x < BSZ) out[threadIdx.x] = 0.f;
}

__global__ void err_reduce(const float* __restrict__ loga, const float* __restrict__ f2,
                           const float* __restrict__ p3dt, float* __restrict__ out) {
    int b = blockIdx.y;
    long base = (long)b * NPTS * NPTS;
    long start = (long)blockIdx.x * 4096 + threadIdx.x;
    float s = 0.f;
#pragma unroll
    for (int j = 0; j < 16; ++j) {
        long idx = start + (long)j * 256;
        int m = (int)(idx >> 10), n = (int)(idx & 1023);
        float w = __expf(loga[base + idx]);
        const float* fm = f2 + (long)b * NPTS * 3 + m * 3;
        const float* pn = p3dt + (long)b * NPTS * 3 + n * 3;
        float dot = fm[0] * pn[0] + fm[1] * pn[1] + fm[2] * pn[2];
        s += w * (1.f - dot);
    }
    __shared__ float sh[256];
    sh[threadIdx.x] = s; __syncthreads();
    for (int o = 128; o > 0; o >>= 1) { if (threadIdx.x < o) sh[threadIdx.x] += sh[threadIdx.x + o]; __syncthreads(); }
    if (threadIdx.x == 0) atomicAdd(out + b, sh[0]);
}

// ---------------------------------------------------------------------------
// host launcher
// ---------------------------------------------------------------------------
extern "C" void kernel_launch(void* const* d_in, const int* in_sizes, int n_in,
                              void* d_out, int out_size, void* d_ws, size_t ws_size,
                              hipStream_t stream) {
    const float* p2d     = (const float*)d_in[0];
    const float* p3d     = (const float*)d_in[1];
    const float* pose    = (const float*)d_in[2];
    const float* enc2d_w = (const float*)d_in[3];
    const float* enc2d_b = (const float*)d_in[4];
    const float* enc3d_w = (const float*)d_in[5];
    const float* enc3d_b = (const float*)d_in[6];
    const float* proj_w  = (const float*)d_in[7];
    const float* proj_b  = (const float*)d_in[8];
    const float* merge_w = (const float*)d_in[9];
    const float* merge_b = (const float*)d_in[10];
    const float* mlp1_w  = (const float*)d_in[11];
    const float* mlp1_b  = (const float*)d_in[12];
    const float* bn_g    = (const float*)d_in[13];
    const float* bn_b    = (const float*)d_in[14];
    const float* mlp2_w  = (const float*)d_in[15];
    const float* mlp2_b  = (const float*)d_in[16];
    float* out = (float*)d_out;

    // ---- workspace carve (floats), total ~44.2 MB ----
    float* w = (float*)d_ws;
    auto alloc = [&](long nfl) { float* p = w; w += nfl; return p; };
    float* f2buf  = alloc(BSZ * NPTS * 3);
    float* p3dt   = alloc(BSZ * NPTS * 3);
    float* n0b    = alloc(BSZ * NPTS);
    float* n1b    = alloc(BSZ * NPTS);
    float* bnmean = alloc(256);
    float* bnvar  = alloc(256);
    const long DS = (long)BSZ * DCH * NPTS;       // 524288
    float* dA0  = alloc(DS); float* dB0 = alloc(DS);
    float* dA1  = alloc(DS); float* dB1 = alloc(DS);
    float* qb   = alloc(DS); float* kb  = alloc(DS);
    float* vb   = alloc(DS); float* attn = alloc(DS);
    float* m2o  = alloc(DS);
    float* ycat = alloc((long)BSZ * 256 * NPTS);  // 1048576
    float* zb   = alloc((long)BSZ * 256 * NPTS);
    float* loga = alloc((long)BSZ * NPTS * NPTS); // 4194304

    // ---- encode ----
    normalize_pts<<<16, 256, 0, stream>>>(p2d, f2buf);
    knn_encode<<<16, 256, 0, stream>>>(f2buf, enc2d_w, enc2d_b, dA0);
    knn_encode<<<16, 256, 0, stream>>>(p3d,   enc3d_w, enc3d_b, dA1);

    auto gemm = [&](const float* Wp, const float* Xp, const float* bp, float* Op,
                    int M, int Kd, long xStride, long oStride, int rowOff) {
        dim3 g(NPTS / 64, M / 16, BSZ);
        wmma_gemm<<<g, 128, 0, stream>>>(Wp, Xp, bp, Op, Kd, xStride, oStride, rowOff);
    };

    float* d0 = dA0; float* d0n = dB0;
    float* d1 = dA1; float* d1n = dB1;

    for (int i = 0; i < 6; ++i) {
        bool cross = (i & 1) != 0;  // NAMES = self,cross,self,cross,self,cross
        const float* pw = proj_w + (long)i * 3 * DCH * DCH;
        const float* pb = proj_b + (long)i * 3 * DCH;
        const float* mw = merge_w + (long)i * DCH * DCH;
        const float* mb = merge_b + (long)i * DCH;
        const float* w1 = mlp1_w + (long)i * 256 * 256;
        const float* b1 = mlp1_b + (long)i * 256;
        const float* gg = bn_g + (long)i * 256;
        const float* bt = bn_b + (long)i * 256;
        const float* w2 = mlp2_w + (long)i * DCH * 256;
        const float* b2 = mlp2_b + (long)i * DCH;

        for (int s = 0; s < 2; ++s) {
            const float* x   = (s == 0) ? d0 : d1;
            const float* src = (s == 0) ? (cross ? d1 : d0) : (cross ? d0 : d1);
            float* dst       = (s == 0) ? d0n : d1n;

            gemm(pw,                 x,   pb,           qb, DCH, DCH, DS / BSZ, DS / BSZ, 0);
            gemm(pw + DCH * DCH,     src, pb + DCH,     kb, DCH, DCH, DS / BSZ, DS / BSZ, 0);
            gemm(pw + 2 * DCH * DCH, src, pb + 2 * DCH, vb, DCH, DCH, DS / BSZ, DS / BSZ, 0);
            flash_attn<<<dim3(NPTS / 64, 4, BSZ), 128, 0, stream>>>(qb, kb, vb, attn);
            gemm(mw, attn, mb, ycat, DCH, DCH, DS / BSZ, 262144, DCH); // rows 128..255
            concat_lower<<<2048, 256, 0, stream>>>(x, ycat);
            gemm(w1, ycat, b1, zb, 256, 256, 262144, 262144, 0);
            bn_stats<<<256, 256, 0, stream>>>(zb, bnmean, bnvar);
            bn_apply_relu<<<4096, 256, 0, stream>>>(zb, gg, bt, bnmean, bnvar);
            gemm(w2, zb, b2, m2o, DCH, 256, 262144, DS / BSZ, 0);
            add_res<<<2048, 256, 0, stream>>>(x, m2o, dst, DS);
        }
        float* t;
        t = d0; d0 = d0n; d0n = t;
        t = d1; d1 = d1n; d1n = t;
    }

    // ---- matching + Sinkhorn ----
    point_norms<<<16, 256, 0, stream>>>(d0, n0b);
    point_norms<<<16, 256, 0, stream>>>(d1, n1b);
    wmma_corr<<<dim3(NPTS / 64, NPTS / 16, BSZ), 128, 0, stream>>>(d0, d1, n0b, n1b, loga);
    for (int it = 0; it < 10; ++it) {
        lse_rows<<<BSZ * NPTS, 256, 0, stream>>>(loga);
        lse_cols<<<BSZ * NPTS, 256, 0, stream>>>(loga);
    }

    // ---- pose transform + final error ----
    transform_points<<<BSZ, 256, 0, stream>>>(pose, p3d, p3dt);
    zero_out<<<1, 32, 0, stream>>>(out);
    err_reduce<<<dim3(256, BSZ), 256, 0, stream>>>(loga, f2buf, p3dt, out);
}
